// AttentionMechanism_11132555231571
// MI455X (gfx1250) — matrix operand order repaired and checked by reading, not compiled, not run on hardware
//
#include <hip/hip_runtime.h>
#include <hip/hip_bf16.h>

// ---------------------------------------------------------------------------
// Problem constants
#define BB    256
#define TT    512
#define DDEC  512
#define DENC  1024
#define DATT  128
#define NF    32
#define KW    31
#define PADW  15
#define MTILE 64                 // rows of (b,t) per workgroup
#define LDSW  40                 // padded LDS row width (u16) for bank spread

typedef __attribute__((ext_vector_type(16))) __bf16   v16bf;
typedef __attribute__((ext_vector_type(8)))  float    v8f;
typedef __attribute__((ext_vector_type(4)))  float    v4f;
typedef __attribute__((ext_vector_type(4)))  unsigned v4u;

union Frag { v16bf v; v4u q[2]; };   // 32B: 16 bf16 halves == 2x b128

__device__ __forceinline__ unsigned short f32_to_bf16(float x) {
    union { float f; unsigned u; } c; c.f = x;
    unsigned b = c.u + 0x7FFFu + ((c.u >> 16) & 1u);   // round-to-nearest-even
    return (unsigned short)(b >> 16);
}
__device__ __forceinline__ float bf16_to_f32(unsigned short h) {
    union { unsigned u; float f; } c; c.u = ((unsigned)h) << 16;
    return c.f;
}

// Error-compensated split of two fp32 into packed bf16 hi-pair + lo-pair.
// hi = truncate-to-bf16 (residual captures the error), packed via v_perm_b32.
__device__ __forceinline__ void split2(float a, float b, unsigned* hp, unsigned* lp) {
    union { float f; unsigned u; } ca, cb, ta, tb, ra, rb;
    ca.f = a; cb.f = b;
    ta.u = ca.u & 0xFFFF0000u;
    tb.u = cb.u & 0xFFFF0000u;
    ra.f = a - ta.f;                       // exact residual
    rb.f = b - tb.f;
    *hp = __builtin_amdgcn_perm(cb.u, ca.u, 0x07060302u);  // {a.hi16, b.hi16}
    *lp = __builtin_amdgcn_perm(rb.u, ra.u, 0x07060302u);
}

// ---------------------------------------------------------------------------
// Prep: Wk (1024x128 fp32) -> bf16 hi/lo, PRE-PACKED in WMMA B-fragment order:
//   idx = ((ntile*32 + kstep)*32 + lane)*16 + h
//   element = Wk[(32*kstep + 8*(lane>>4) + (h&7) + 16*(h>>3))*128 + ntile*16 + (lane&15)]
__global__ void prep_wk(const float* __restrict__ Wk,
                        unsigned short* __restrict__ hiP,
                        unsigned short* __restrict__ loP) {
    int i = blockIdx.x * blockDim.x + threadIdx.x;
    if (i >= DENC * DATT) return;
    int h    = i & 15;
    int lane = (i >> 4) & 31;
    int s    = (i >> 9) & 31;
    int nt   = i >> 14;
    int k    = 32 * s + 8 * (lane >> 4) + (h & 7) + 16 * (h >> 3);
    int n    = nt * 16 + (lane & 15);
    float x  = Wk[k * DATT + n];
    unsigned short hh = f32_to_bf16(x);
    hiP[i] = hh;
    loP[i] = f32_to_bf16(x - bf16_to_f32(hh));
}

// Prep: packed location-GEMM B matrix: W2[k,a] = sum_f conv_w[f,k]*Wloc[f,a],
// padded to K=32 (row 31 = 0), in B-fragment order (single k-step):
//   idx = (ntile*32 + lane)*16 + h ; k = 8*(lane>>4)+(h&7)+16*(h>>3) ; n = ntile*16+(lane&15)
__global__ void prep_w2pack(const float* __restrict__ conv_w,
                            const float* __restrict__ Wloc,
                            unsigned short* __restrict__ hiP,
                            unsigned short* __restrict__ loP) {
    int i = blockIdx.x * blockDim.x + threadIdx.x;
    if (i >= 8 * 32 * 16) return;
    int h    = i & 15;
    int lane = (i >> 4) & 31;
    int nt   = i >> 9;
    int k    = 8 * (lane >> 4) + (h & 7) + 16 * (h >> 3);
    int n    = nt * 16 + (lane & 15);
    float x  = 0.f;
    if (k < KW)
        for (int f = 0; f < NF; ++f)
            x = fmaf(conv_w[f * KW + k], Wloc[f * DATT + n], x);
    unsigned short hh = f32_to_bf16(x);
    hiP[i] = hh;
    loP[i] = f32_to_bf16(x - bf16_to_f32(hh));
}

// biasA[a] = sum_f conv_b[f]*Wloc[f,a]
__global__ void prep_bias(const float* __restrict__ conv_b,
                          const float* __restrict__ Wloc,
                          float* __restrict__ biasA) {
    int a = threadIdx.x;
    float acc = 0.f;
    for (int f = 0; f < NF; ++f)
        acc = fmaf(conv_b[f], Wloc[f * DATT + a], acc);
    biasA[a] = acc;
}

// q = query @ Wq  (256x512 @ 512x128)
__global__ void qproj_kernel(const float* __restrict__ query,
                             const float* __restrict__ Wq,
                             float* __restrict__ q) {
    int b = blockIdx.x, a = threadIdx.x;
    float acc = 0.f;
    for (int kk = 0; kk < DDEC; ++kk)
        acc = fmaf(query[b * DDEC + kk], Wq[kk * DATT + a], acc);
    q[b * DATT + a] = acc;
}

// ---------------------------------------------------------------------------
// Fused energy kernel: energy[b,t] = tanh(qproj + E@Wk + loc) @ Wscore
// WG = 64 rows x 128 att cols, 8 waves; wave w owns cols [16w,16w+16).
// K loop: 12 wmma/wave/step on E@Wk (bf16 hi/lo).  Location conv folded into
// the SAME accumulators as a 64x32 @ 32x128 WMMA GEMM (taps tile built once).
__global__ __launch_bounds__(256) void energy_kernel(
        const float* __restrict__ E,
        const unsigned short* __restrict__ WkHiP,
        const unsigned short* __restrict__ WkLoP,
        const unsigned short* __restrict__ W2HiP,
        const unsigned short* __restrict__ W2LoP,
        const float* __restrict__ qproj,
        const float* __restrict__ biasA,
        const float* __restrict__ attn,
        const float* __restrict__ Wscore,
        float* __restrict__ energy) {
    __shared__ __align__(16) unsigned short sAhi[MTILE][LDSW];
    __shared__ __align__(16) unsigned short sAlo[MTILE][LDSW];
    __shared__ __align__(16) unsigned short sLhi[MTILE][LDSW];
    __shared__ __align__(16) unsigned short sLlo[MTILE][LDSW];
    __shared__ float sAtt[96];               // 64+31 taps (+1 pad, zero-safe)
    __shared__ float sEnergy[MTILE];

    const int tid  = threadIdx.x;
    const int wave = tid >> 5;
    const int lane = tid & 31;
    const int r0   = blockIdx.x * MTILE;    // flat (b*T + t) row base
    const int bb   = r0 >> 9;               // / TT  (MTILE divides TT)
    const int t0   = r0 & (TT - 1);

    if (tid < 96) {
        int tg = t0 - PADW + tid;
        sAtt[tid] = (tg >= 0 && tg < TT) ? attn[bb * TT + tg] : 0.0f;
    }
    if (tid < MTILE) sEnergy[tid] = 0.0f;

    const int hi   = lane >> 4;             // which 16-lane half
    const int hl   = lane & 15;
    const int ncol = wave * 16 + hl;        // this lane's att-dim column

    // staging role: thread owns 8 consecutive elements of a 64x32 tile
    const int srow = tid >> 2;              // 0..63
    const int scol = (tid & 3) * 8;         // 0,8,16,24
    const float* __restrict__ Erow = E + (size_t)(r0 + srow) * DENC + scol;

    __syncthreads();
    // ---- build location-taps tile once: sL[m][k] = sAtt[m+k] as bf16 hi/lo
    {
        unsigned h0, h1, h2, h3, l0, l1, l2, l3;
        const float* tp = &sAtt[srow + scol];
        split2(tp[0], tp[1], &h0, &l0);
        split2(tp[2], tp[3], &h1, &l1);
        split2(tp[4], tp[5], &h2, &l2);
        split2(tp[6], tp[7], &h3, &l3);
        v4u hq = { h0, h1, h2, h3 };
        v4u lq = { l0, l1, l2, l3 };
        *(v4u*)&sLhi[srow][scol] = hq;
        *(v4u*)&sLlo[srow][scol] = lq;
    }
    // (K-loop's leading barrier orders these writes before any reads)

    Frag ahi, alo, bhi, blo;
    v8f acc[4] = {};

    for (int ks = 0; ks < DENC; ks += 32) {
        __syncthreads();
        // ---- stage: 2x global b128 loads, perm-pack hi/lo, 2x ds b128 stores
        {
            const v4f* src = (const v4f*)(Erow + ks);
            v4f x0 = src[0];
            v4f x1 = src[1];
            unsigned h0, h1, h2, h3, l0, l1, l2, l3;
            split2(x0.x, x0.y, &h0, &l0);
            split2(x0.z, x0.w, &h1, &l1);
            split2(x1.x, x1.y, &h2, &l2);
            split2(x1.z, x1.w, &h3, &l3);
            v4u hq = { h0, h1, h2, h3 };
            v4u lq = { l0, l1, l2, l3 };
            *(v4u*)&sAhi[srow][scol] = hq;
            *(v4u*)&sAlo[srow][scol] = lq;
            if (ks + 32 < DENC)
                __builtin_prefetch(Erow + ks + 32, 0, 3);   // global_prefetch_b8
        }
        __syncthreads();

        // ---- B fragment: pre-packed, 32 contiguous bytes per lane
        {
            const int s = ks >> 5;
            const size_t fo = ((size_t)((wave * 32 + s) * 32 + lane)) << 4;
            const v4u* pH = (const v4u*)(WkHiP + fo);
            const v4u* pL = (const v4u*)(WkLoP + fo);
            bhi.q[0] = pH[0]; bhi.q[1] = pH[1];
            blo.q[0] = pL[0]; blo.q[1] = pL[1];
        }

        // ---- 4 M-subtiles reuse the B fragment
        #pragma unroll
        for (int mt = 0; mt < 4; ++mt) {
            const unsigned short* ar = &sAhi[mt * 16 + hl][0];
            const unsigned short* al = &sAlo[mt * 16 + hl][0];
            // A layout: halves 0-7 = K[8*hi..8*hi+7], halves 8-15 = K[16+8*hi..]
            ahi.q[0] = *(const v4u*)(ar + 8 * hi);
            ahi.q[1] = *(const v4u*)(ar + 16 + 8 * hi);
            alo.q[0] = *(const v4u*)(al + 8 * hi);
            alo.q[1] = *(const v4u*)(al + 16 + 8 * hi);
            acc[mt] = __builtin_amdgcn_wmma_f32_16x16x32_bf16(false, ahi.v, false, bhi.v, (short)0, acc[mt], false, false);
            acc[mt] = __builtin_amdgcn_wmma_f32_16x16x32_bf16(false, ahi.v, false, blo.v, (short)0, acc[mt], false, false);
            acc[mt] = __builtin_amdgcn_wmma_f32_16x16x32_bf16(false, alo.v, false, bhi.v, (short)0, acc[mt], false, false);
        }
    }

    // ---- location conv as WMMA: acc += Taps(64x32) @ W2(32x128)
    {
        Frag w2h, w2l;
        const size_t fo = ((size_t)(wave * 32 + lane)) << 4;
        const v4u* pH = (const v4u*)(W2HiP + fo);
        const v4u* pL = (const v4u*)(W2LoP + fo);
        w2h.q[0] = pH[0]; w2h.q[1] = pH[1];
        w2l.q[0] = pL[0]; w2l.q[1] = pL[1];
        #pragma unroll
        for (int mt = 0; mt < 4; ++mt) {
            const unsigned short* lr = &sLhi[mt * 16 + hl][0];
            const unsigned short* ll = &sLlo[mt * 16 + hl][0];
            ahi.q[0] = *(const v4u*)(lr + 8 * hi);
            ahi.q[1] = *(const v4u*)(lr + 16 + 8 * hi);
            alo.q[0] = *(const v4u*)(ll + 8 * hi);
            alo.q[1] = *(const v4u*)(ll + 16 + 8 * hi);
            acc[mt] = __builtin_amdgcn_wmma_f32_16x16x32_bf16(false, ahi.v, false, w2h.v, (short)0, acc[mt], false, false);
            acc[mt] = __builtin_amdgcn_wmma_f32_16x16x32_bf16(false, ahi.v, false, w2l.v, (short)0, acc[mt], false, false);
            acc[mt] = __builtin_amdgcn_wmma_f32_16x16x32_bf16(false, alo.v, false, w2h.v, (short)0, acc[mt], false, false);
        }
    }

    // ---- epilogue: + q + bias, tanh, * Wscore, reduce over 128 cols
    const float qb = qproj[bb * DATT + ncol] + biasA[ncol];
    const float sc = Wscore[ncol];
    const int mbase = hi * 8;               // C layout: VGPR j -> M = j + 8*hi
    #pragma unroll
    for (int mt = 0; mt < 4; ++mt) {
        float esum[8];
        #pragma unroll
        for (int j = 0; j < 8; ++j) {
            float val = tanhf(acc[mt][j] + qb) * sc;
            #pragma unroll
            for (int off = 8; off >= 1; off >>= 1)
                val += __shfl_xor(val, off, 16);    // sum over 16 cols
            esum[j] = val;
        }
        if (hl == 0) {
            #pragma unroll
            for (int j = 0; j < 8; ++j)
                atomicAdd(&sEnergy[mt * 16 + mbase + j], esum[j]);
        }
    }
    __syncthreads();
    if (tid < MTILE) energy[r0 + tid] = sEnergy[tid];
}

// ---------------------------------------------------------------------------
// Softmax over T per batch row -> new_w
__global__ __launch_bounds__(256) void softmax_kernel(const float* __restrict__ energy,
                                                      float* __restrict__ outw) {
    __shared__ float red[256];
    int b = blockIdx.x, tid = threadIdx.x;
    float e0 = energy[b * TT + tid];
    float e1 = energy[b * TT + 256 + tid];
    float m = fmaxf(e0, e1);
    red[tid] = m; __syncthreads();
    for (int s = 128; s > 0; s >>= 1) {
        if (tid < s) red[tid] = fmaxf(red[tid], red[tid + s]);
        __syncthreads();
    }
    m = red[0]; __syncthreads();
    float x0 = __expf(e0 - m), x1 = __expf(e1 - m);
    red[tid] = x0 + x1; __syncthreads();
    for (int s = 128; s > 0; s >>= 1) {
        if (tid < s) red[tid] += red[tid + s];
        __syncthreads();
    }
    float inv = 1.0f / red[0];
    outw[b * TT + tid]       = x0 * inv;
    outw[b * TT + 256 + tid] = x1 * inv;
}

// ctx1[b,e] = sum_t w[b,t] * E[b,t,e]   (second streaming pass over E, b128)
__global__ __launch_bounds__(256) void ctx1_kernel(const float* __restrict__ E,
                                                   const float* __restrict__ w,
                                                   float* __restrict__ ctx1) {
    int b = blockIdx.x, tid = threadIdx.x;
    v4f acc = { 0.f, 0.f, 0.f, 0.f };
    for (int t = 0; t < TT; ++t) {
        float wv = w[b * TT + t];
        const v4f* row = (const v4f*)(E + ((size_t)b * TT + t) * DENC) + tid;
        acc += wv * row[0];
    }
    ((v4f*)(ctx1 + (size_t)b * DENC))[tid] = acc;
}

// context[b,d] = ctx1[b,:] @ Wv   (256x1024 @ 1024x512, Wv L2-resident)
__global__ __launch_bounds__(512) void ctxout_kernel(const float* __restrict__ ctx1,
                                                     const float* __restrict__ Wv,
                                                     float* __restrict__ out) {
    int b = blockIdx.x, d = threadIdx.x;
    float acc = 0.f;
    for (int e = 0; e < DENC; ++e)
        acc = fmaf(ctx1[b * DENC + e], Wv[e * DDEC + d], acc);
    out[b * DDEC + d] = acc;
}

// ---------------------------------------------------------------------------
extern "C" void kernel_launch(void* const* d_in, const int* in_sizes, int n_in,
                              void* d_out, int out_size, void* d_ws, size_t ws_size,
                              hipStream_t stream) {
    const float* query  = (const float*)d_in[0];   // (B, 512)
    const float* E      = (const float*)d_in[1];   // (B, T, 1024)
    const float* attn   = (const float*)d_in[2];   // (B, T)
    const float* Wq     = (const float*)d_in[3];   // (512, 128)
    const float* Wk     = (const float*)d_in[4];   // (1024, 128)
    const float* Wv     = (const float*)d_in[5];   // (1024, 512)
    const float* Wloc   = (const float*)d_in[6];   // (32, 128)
    const float* conv_w = (const float*)d_in[7];   // (32, 1, 31)
    const float* conv_b = (const float*)d_in[8];   // (32,)
    const float* Wscore = (const float*)d_in[9];   // (128, 1)

    float* out_ctx = (float*)d_out;                       // (B, 512)
    float* out_w   = (float*)d_out + BB * DDEC;           // (B, T)

    // workspace carve-up (~2.2 MB)
    float* wsf    = (float*)d_ws;
    float* q      = wsf;                    // 32768
    float* biasA  = wsf + 32768;            // 128
    float* energy = wsf + 32896;            // 131072
    float* ctx1   = wsf + 163968;           // 262144
    unsigned short* wkhi = (unsigned short*)(wsf + 426112);  // 131072 u16
    unsigned short* wklo = wkhi + DENC * DATT;               // 131072 u16
    unsigned short* w2hi = wklo + DENC * DATT;               // 8192 u16
    unsigned short* w2lo = w2hi + 8192;                      // 8192 u16

    prep_wk<<<(DENC * DATT + 255) / 256, 256, 0, stream>>>(Wk, wkhi, wklo);
    prep_w2pack<<<32, 256, 0, stream>>>(conv_w, Wloc, w2hi, w2lo);
    prep_bias<<<1, DATT, 0, stream>>>(conv_b, Wloc, biasA);
    qproj_kernel<<<BB, DATT, 0, stream>>>(query, Wq, q);

    energy_kernel<<<(BB * TT) / MTILE, 256, 0, stream>>>(
        E, wkhi, wklo, w2hi, w2lo, q, biasA, attn, Wscore, energy);

    softmax_kernel<<<BB, 256, 0, stream>>>(energy, out_w);
    ctx1_kernel<<<BB, 256, 0, stream>>>(E, out_w, ctx1);
    ctxout_kernel<<<BB, DDEC, 0, stream>>>(ctx1, Wv, out_ctx);
}
